// AttentionWithoutParams_15994458210630
// MI455X (gfx1250) — matrix operand order repaired and checked by reading, not compiled, not run on hardware
//
#include <hip/hip_runtime.h>
#include <hip/hip_bf16.h>
#include <math.h>
#include <stdint.h>

typedef __bf16 bf16_t;
typedef __attribute__((ext_vector_type(16))) __bf16 v16bf;
typedef __attribute__((ext_vector_type(8)))  float  v8f;
typedef __attribute__((ext_vector_type(4)))  unsigned int u32x4;

// TDM descriptor vector types
typedef __attribute__((ext_vector_type(4))) unsigned int v4u32;
typedef __attribute__((ext_vector_type(8))) int          v8i32;
typedef __attribute__((ext_vector_type(4))) int          v4i32;

#if defined(__has_builtin)
# if __has_builtin(__builtin_amdgcn_tensor_load_to_lds)
#  define HAVE_TDM 1
# endif
#endif

#define BATCH   2
#define N_LAT   1024
#define N_CTX   4096
#define DMODEL  1024
#define HEADS   16
#define HD      64
#define QT      2    // 16-row q tiles per wave (32 q rows/wave)
#define JTILE   64   // keys per TDM tile

// bf16 fragment = two 16-byte contiguous LDS reads (no per-wave conversion)
union Frag {
  v16bf v;
  struct { u32x4 lo, hi; } q;
};
static __device__ inline v16bf load_frag(const bf16_t* p) {
  Frag f;
  f.q.lo = *(const u32x4*)(p);
  f.q.hi = *(const u32x4*)(p + 8);
  return f.v;
}

#ifdef HAVE_TDM
// 2D TDM tile: 64 rows x 64 f32, row stride 1024 f32, global -> LDS
static __device__ inline void tdm_load_tile(unsigned int lds_addr, const float* gptr) {
  const uint64_t gaddr = (uint64_t)(uintptr_t)gptr;
  // group0: count=1 | lds_addr | global_addr[56:0] | type=2 (bits 127:126)
  v4u32 g0 = { 1u,
               lds_addr,
               (unsigned int)(gaddr & 0xFFFFFFFFu),
               (unsigned int)((gaddr >> 32) & 0x01FFFFFFu) | 0x80000000u };
  // group1: data_size=4B | tensor_dim0=64 | tensor_dim1=64 | tile 64x64 | stride0=1024
  v8i32 g1 = { (int)(2u << 16),      // workgroup_mask=0, data_size=2 (4B)
               (int)(64u << 16),     // tensor_dim0[15:0]=64
               (int)(64u << 16),     // tensor_dim1[15:0]=64
               (int)(64u << 16),     // tile_dim0=64
               64,                   // tile_dim1=64, tile_dim2=0
               1024,                 // tensor_dim0_stride low 32
               0, 0 };
  v4i32 gz = { 0, 0, 0, 0 };
# if __clang_major__ >= 23
  v8i32 gz8 = { 0, 0, 0, 0, 0, 0, 0, 0 };
  __builtin_amdgcn_tensor_load_to_lds(g0, g1, gz, gz, gz8, 0);
# else
  __builtin_amdgcn_tensor_load_to_lds(g0, g1, gz, gz, 0);
# endif
}
#endif

// ---------------------------------------------------------------------------
// Flash attention. One block = 8 waves sharing one (b, h); per iteration a
// 64x64 f32 context tile (context is both K and V) is TDM-DMA'd into a
// double-buffered LDS staging area (next DMA overlapped with compute), then
// converted ONCE by the whole block into bf16 K (row-major) and V^T
// (transposed) LDS copies. Waves load WMMA fragments as raw ds_load_b128.
//   S^T = K * Q^T      (softmax axis j on the VGPR axis of the C layout)
//   O^T = V^T * P^T    (online-softmax rescaled accumulate)
// ---------------------------------------------------------------------------
__global__ __launch_bounds__(256)
void fa_fwd(const float* __restrict__ x, const float* __restrict__ ctx,
            bf16_t* __restrict__ o_ws)
{
#ifdef HAVE_TDM
  __shared__ __align__(16) float  f32buf[2][JTILE * 64];  // 2 x 16 KB TDM dest
#endif
  __shared__ __align__(16) bf16_t kbuf[JTILE * 64];       // 8 KB, (j, d) row-major
  __shared__ __align__(16) bf16_t vtbuf[64 * JTILE];      // 8 KB, (d, j) transposed

  const int lane = threadIdx.x & 31;
  const int wv   = threadIdx.x >> 5;

  // block -> (b, h, 256-row q chunk)
  const int qch = blockIdx.x & 3;
  const int bh  = blockIdx.x >> 2;
  const int h   = bh & (HEADS - 1);
  const int b   = bh >> 4;

  const int n  = lane & 15;           // fragment column / row index
  const int hi = lane >> 4;           // lane half
  const float scale = 0.125f;         // 1/sqrt(64)

  const float* ctx_base = ctx + (size_t)b * N_CTX * DMODEL + h * HD;

  // ---- Q^T B-fragments (d x i), loaded once: [qt][c], c = d-chunk of 32
  v16bf qb[QT][2];
  #pragma unroll
  for (int qt = 0; qt < QT; ++qt) {
    const int qrow = qch * 256 + wv * 32 + qt * 16 + n;
    const float* xp = x + ((size_t)b * N_LAT + qrow) * DMODEL + h * HD;
    #pragma unroll
    for (int c = 0; c < 2; ++c) {
      const int kb = c * 32 + hi * 16;
      #pragma unroll
      for (int e = 0; e < 16; ++e)
        qb[qt][c][e] = (bf16_t)xp[kb + e];
    }
  }

  float m_run[QT], l_run[QT];
  v8f o_acc[QT][4];
  #pragma unroll
  for (int qt = 0; qt < QT; ++qt) {
    m_run[qt] = -1e30f; l_run[qt] = 0.0f;
    #pragma unroll
    for (int mt = 0; mt < 4; ++mt) o_acc[qt][mt] = v8f{};
  }

#ifdef HAVE_TDM
  const unsigned int lds_f32_base = (unsigned int)(uintptr_t)(&f32buf[0][0]);
  // prologue: first tile DMA in flight before the loop
  if (wv == 0) tdm_load_tile(lds_f32_base, ctx_base);
#endif

  for (int it = 0, jb = 0; jb < N_CTX; jb += JTILE, ++it) {
    const int cur = it & 1;

#ifdef HAVE_TDM
    if (wv == 0) __builtin_amdgcn_s_wait_tensorcnt(0);  // current tile landed
#endif
    __syncthreads();   // tile visible; kbuf/vtbuf free (prev compute done)

#ifdef HAVE_TDM
    if (wv == 0 && jb + JTILE < N_CTX)  // overlap next DMA with this compute
      tdm_load_tile(lds_f32_base + (unsigned int)((1 - cur) * JTILE * 64 * 4),
                    ctx_base + (size_t)(jb + JTILE) * DMODEL);
    // cooperative one-pass conversion: f32 tile -> bf16 K and V^T copies
    #pragma unroll 4
    for (int t = threadIdx.x; t < JTILE * 64; t += 256) {
      const bf16_t v = (bf16_t)f32buf[cur][t];
      kbuf[t] = v;
      vtbuf[(t & 63) * JTILE + (t >> 6)] = v;
    }
#else
    {
      const float* src = ctx_base + (size_t)jb * DMODEL;
      #pragma unroll 4
      for (int t = threadIdx.x; t < JTILE * 64; t += 256) {
        const bf16_t v = (bf16_t)src[(size_t)(t >> 6) * DMODEL + (t & 63)];
        kbuf[t] = v;
        vtbuf[(t & 63) * JTILE + (t >> 6)] = v;
      }
    }
#endif
    __syncthreads();   // bf16 copies ready

    #pragma unroll
    for (int sub = 0; sub < 2; ++sub) {
      const int j0 = sub * 32;

      // K A-fragments (j rows x d): contiguous b128 pairs from kbuf
      v16bf ka[2][2];
      #pragma unroll
      for (int jt = 0; jt < 2; ++jt)
        #pragma unroll
        for (int c = 0; c < 2; ++c)
          ka[jt][c] = load_frag(&kbuf[(j0 + jt * 16 + n) * 64 + c * 32 + hi * 16]);

      // V^T A-fragments (d rows x j): contiguous b128 pairs from vtbuf
      v16bf va[4];
      #pragma unroll
      for (int mt = 0; mt < 4; ++mt)
        va[mt] = load_frag(&vtbuf[(mt * 16 + n) * JTILE + j0 + hi * 16]);

      #pragma unroll
      for (int qt = 0; qt < QT; ++qt) {
        // S^T tiles: rows j (VGPR axis), cols i (lane axis)
        v8f s0 = v8f{}, s1 = v8f{};
        s0 = __builtin_amdgcn_wmma_f32_16x16x32_bf16(false, ka[0][0], false, qb[qt][0], (short)0, s0, false, false);
        s0 = __builtin_amdgcn_wmma_f32_16x16x32_bf16(false, ka[0][1], false, qb[qt][1], (short)0, s0, false, false);
        s1 = __builtin_amdgcn_wmma_f32_16x16x32_bf16(false, ka[1][0], false, qb[qt][0], (short)0, s1, false, false);
        s1 = __builtin_amdgcn_wmma_f32_16x16x32_bf16(false, ka[1][1], false, qb[qt][1], (short)0, s1, false, false);

        // Online softmax: in-lane over 16 j's, one xor-16 shuffle merges halves
        float mloc = -1e30f;
        #pragma unroll
        for (int r = 0; r < 8; ++r) {
          s0[r] *= scale; s1[r] *= scale;
          mloc = fmaxf(mloc, fmaxf(s0[r], s1[r]));
        }
        mloc = fmaxf(mloc, __shfl_xor(mloc, 16, 32));
        const float m_new = fmaxf(m_run[qt], mloc);
        const float alpha = __expf(m_run[qt] - m_new);
        float lloc = 0.0f;
        #pragma unroll
        for (int r = 0; r < 8; ++r) {
          s0[r] = __expf(s0[r] - m_new);
          s1[r] = __expf(s1[r] - m_new);
          lloc += s0[r] + s1[r];
        }
        lloc += __shfl_xor(lloc, 16, 32);
        l_run[qt] = l_run[qt] * alpha + lloc;
        m_run[qt] = m_new;

        #pragma unroll
        for (int mt = 0; mt < 4; ++mt)
          #pragma unroll
          for (int r = 0; r < 8; ++r)
            o_acc[qt][mt][r] *= alpha;

        // Relayout P^T: C-frag (M over VGPRs) -> B-frag (K sequential per half)
        v16bf pb;
        #pragma unroll
        for (int r = 0; r < 8; ++r) {
          const float p0 = __shfl_xor(s0[r], 16, 32);
          const float p1 = __shfl_xor(s1[r], 16, 32);
          pb[r]     = (bf16_t)(hi ? p1    : s0[r]);
          pb[8 + r] = (bf16_t)(hi ? s1[r] : p0);
        }

        // O^T accumulate
        #pragma unroll
        for (int mt = 0; mt < 4; ++mt)
          o_acc[qt][mt] = __builtin_amdgcn_wmma_f32_16x16x32_bf16(false, va[mt], false, pb, (short)0, o_acc[qt][mt], false, false);
      }
    }
  }

  // ---- normalize, store O as (b, i, d_model) bf16
  #pragma unroll
  for (int qt = 0; qt < QT; ++qt) {
    const float linv = 1.0f / l_run[qt];
    const int qrow = qch * 256 + wv * 32 + qt * 16 + n;
    bf16_t* op = o_ws + ((size_t)b * N_LAT + qrow) * DMODEL + h * HD;
    #pragma unroll
    for (int mt = 0; mt < 4; ++mt)
      #pragma unroll
      for (int r = 0; r < 8; ++r)
        op[mt * 16 + hi * 8 + r] = (bf16_t)(o_acc[qt][mt][r] * linv);
  }
}

// ---------------------------------------------------------------------------
// Output projection: Y[r,o] = sum_k O[r,k] * W[o,k] + bias[o]
// One 16x16 output tile per wave, K=1024 in 32-wide WMMA steps.
// ---------------------------------------------------------------------------
__global__ __launch_bounds__(256)
void proj_gemm(const bf16_t* __restrict__ xo, const float* __restrict__ W,
               const float* __restrict__ bias, float* __restrict__ out)
{
  const int lane = threadIdx.x & 31;
  const int wv   = threadIdx.x >> 5;
  const int tile = blockIdx.x * 8 + wv;
  const int ot   = tile & 63;          // 64 col tiles (o)
  const int rt   = tile >> 6;          // 128 row tiles (r)

  const int n  = lane & 15;
  const int hi = lane >> 4;

  const bf16_t* ap = xo + (size_t)(rt * 16 + n) * DMODEL;   // A row = r
  const float*  bp = W  + (size_t)(ot * 16 + n) * DMODEL;   // B col o -> W row o

  v8f acc = v8f{};
  for (int kb = 0; kb < DMODEL; kb += 32) {
    __builtin_prefetch(bp + kb + 64, 0, 1);   // global_prefetch_b8 on W stream
    v16bf a, bf;
    const int ak = kb + hi * 16;
    const int bk = kb + hi * 16;
    #pragma unroll
    for (int e = 0; e < 8; ++e) {
      a[e]     = ap[ak + e];
      a[8 + e] = ap[ak + 8 + e];
    }
    #pragma unroll
    for (int e = 0; e < 16; ++e)
      bf[e] = (bf16_t)bp[bk + e];
    acc = __builtin_amdgcn_wmma_f32_16x16x32_bf16(false, a, false, bf, (short)0, acc, false, false);
  }

  const float bv = bias[ot * 16 + n];
  #pragma unroll
  for (int r = 0; r < 8; ++r) {
    const int row = rt * 16 + hi * 8 + r;
    out[(size_t)row * DMODEL + ot * 16 + n] = acc[r] + bv;
  }
}

// ---------------------------------------------------------------------------
extern "C" void kernel_launch(void* const* d_in, const int* in_sizes, int n_in,
                              void* d_out, int out_size, void* d_ws, size_t ws_size,
                              hipStream_t stream) {
  (void)in_sizes; (void)n_in; (void)out_size; (void)ws_size;
  const float* x    = (const float*)d_in[0];
  const float* ctx  = (const float*)d_in[1];
  const float* Wout = (const float*)d_in[2];
  const float* bout = (const float*)d_in[3];
  // d_in[4] = heads (fixed at 16, baked into the kernel)

  bf16_t* o_ws = (bf16_t*)d_ws;   // 2048*1024 bf16 = 4 MB scratch

  // 2*16*4 = 128 blocks, 8 waves each (one (b,h) + 256-row q chunk per block)
  fa_fwd<<<128, 256, 0, stream>>>(x, ctx, o_ws);
  // 128*64 = 8192 tiles, 8 waves/block
  proj_gemm<<<1024, 256, 0, stream>>>(o_ws, Wout, bout, (float*)d_out);
}
